// CustomLoss_29746943492152
// MI455X (gfx1250) — compile-verified, matrix operand order checked by reading, not run on hardware
//
#include <hip/hip_runtime.h>
#include <hip/hip_bf16.h>
#include <math.h>

typedef __attribute__((ext_vector_type(16))) _Float16 v16h;
typedef __attribute__((ext_vector_type(8)))  float    v8f;

#define BB 32
#define PP 8732
#define CC 81
#define TILES ((PP + 15) / 16)   /* 546 */
#define NPR 3                    /* NEG_POS_RATIO */

// d_ws layout:
//   float conf_masked[BB*PP]
//   int   pos_count[BB]
//   float accum[3]   // 0: loc_sum, 1: pos_conf_sum, 2: neg_conf_sum

__global__ void init_kernel(int* pos_count, float* accum) {
  int t = threadIdx.x;
  if (t < BB) pos_count[t] = 0;
  if (t < 3)  accum[t] = 0.f;
}

__global__ __launch_bounds__(32)
void conf_kernel(const float* __restrict__ act_deltas,
                 const float* __restrict__ act_labels,
                 const float* __restrict__ pred_deltas,
                 const float* __restrict__ pred_labels,
                 float* __restrict__ conf_masked,
                 int*   __restrict__ pos_count,
                 float* __restrict__ accum)
{
  const int lane = threadIdx.x;
  const int bm = lane & 15;            // box within tile that this lane loads for
  const int hi = lane >> 4;            // which K-half (WMMA A-layout)
  const int b  = blockIdx.y;
  const int p0 = blockIdx.x * 16;
  const int p  = p0 + bm;
  const bool valid = (p < PP);
  const int  pcl = valid ? p : (PP - 1);
  const size_t rowo = ((size_t)b * PP + pcl) * CC;
  const float* Lrow = pred_labels + rowo;
  const float* Arow = act_labels  + rowo;

  // Load the 48 classes this lane owns in WMMA A-matrix order:
  // chunk ck (K-base 32*ck): halves 0..7 = classes cb+8*hi+j, halves 8..15 = cb+16+8*hi+j
  float r[48];
  float pmax = -1e30f, pdot = 0.f;
#pragma unroll
  for (int ck = 0; ck < 3; ++ck) {
    const int cb = 32 * ck;
#pragma unroll
    for (int g = 0; g < 2; ++g) {
      const int s = cb + 16 * g + 8 * hi;
#pragma unroll
      for (int j = 0; j < 8; ++j) {
        const int c = s + j;
        const bool ok = valid && (c < CC);
        float x = ok ? Lrow[c] : -1e30f;   // sentinel: exp underflows to 0
        float a = ok ? Arow[c] : 0.f;
        r[ck * 16 + g * 8 + j] = x;
        pmax = fmaxf(pmax, x);
        pdot = fmaf(a, x, pdot);           // one-hot dot with logits
      }
    }
  }
  // combine the two K-halves of each box (lanes bm and bm+16)
  float cmax = fmaxf(pmax, __shfl_xor(pmax, 16, 32));
  float cdot = pdot + __shfl_xor(pdot, 16, 32);

  // Sum_k exp(x - max) on the matrix unit: D = A(exp) x B(ones), f32 accumulate
  v16h ones;
#pragma unroll
  for (int j = 0; j < 16; ++j) ones[j] = (_Float16)1.0f;

  v8f acc = {0.f, 0.f, 0.f, 0.f, 0.f, 0.f, 0.f, 0.f};
#pragma unroll
  for (int ck = 0; ck < 3; ++ck) {
    v16h av;
#pragma unroll
    for (int j = 0; j < 16; ++j)
      av[j] = (_Float16)__expf(r[ck * 16 + j] - cmax);
    acc = __builtin_amdgcn_wmma_f32_16x16x32_f16(false, av, false, ones,
                                                 (short)0, acc, false, false);
  }

  // C/D layout: lanes 0-15 hold boxes 0..7 (M=v in acc[v]); lanes 16-31 hold boxes 8..15.
  // Writers: lanes 0..7 -> box lane (acc[lane]); lanes 16..23 -> box 8+(lane-16) (acc[lane-16]).
  const bool isw  = (lane < 8) || (lane >= 16 && lane < 24);
  const int  widx = (lane < 8) ? lane : (lane - 16);
  const int  wbox = (lane < 8) ? lane : (lane - 8);
  float es = 0.f;
#pragma unroll
  for (int v = 0; v < 8; ++v) es = (widx == v) ? acc[v] : es;

  const int src = (wbox >= 0 && wbox < 16) ? wbox : 0;  // max/dot live in lanes 0..15
  const float mb = __shfl(cmax, src, 32);
  const float db = __shfl(cdot, src, 32);
  const float conf = __logf(es) + mb - db;              // lse - dot

  float loc_c = 0.f, pconf_c = 0.f;
  int posc = 0;
  const int wp = p0 + wbox;
  if (isw && wp < PP) {
    const size_t o4 = ((size_t)b * PP + wp) * 4;        // 16B aligned
    const float4 ad = *(const float4*)(act_deltas  + o4);
    const float4 pd = *(const float4*)(pred_deltas + o4);
    const bool pos = (ad.x != 0.f) || (ad.y != 0.f) || (ad.z != 0.f) || (ad.w != 0.f);
    float h = 0.f;
    float dx[4] = {pd.x - ad.x, pd.y - ad.y, pd.z - ad.z, pd.w - ad.w};
#pragma unroll
    for (int i2 = 0; i2 < 4; ++i2) {
      const float d = dx[i2], adf = fabsf(d);
      h += (adf <= 1.f) ? 0.5f * d * d : (adf - 0.5f);
    }
    h *= 0.25f;                                          // Huber mean over last axis
    conf_masked[(size_t)b * PP + wp] = pos ? -__builtin_inff() : conf;
    if (pos) { posc = 1; loc_c = h; pconf_c = conf; }
  }
  // wave reduction -> 3 atomics per wave
#pragma unroll
  for (int off = 16; off > 0; off >>= 1) {
    loc_c   += __shfl_xor(loc_c,   off, 32);
    pconf_c += __shfl_xor(pconf_c, off, 32);
    posc    += __shfl_xor(posc,    off, 32);
  }
  if (lane == 0) {
    if (posc)           atomicAdd(&pos_count[b], posc);
    if (loc_c != 0.f)   atomicAdd(&accum[0], loc_c);
    if (pconf_c != 0.f) atomicAdd(&accum[1], pconf_c);
  }
}

// Exact top-k SUM per row via 4-round 256-bin radix descent on monotone float keys.
__global__ __launch_bounds__(256)
void mine_kernel(const float* __restrict__ conf_masked,
                 const int*   __restrict__ pos_count,
                 float* __restrict__ accum)
{
  __shared__ float    vals[PP];
  __shared__ unsigned hist[256];
  __shared__ float    hsum[256];
  __shared__ unsigned s_prefix, s_krem;
  __shared__ float    s_sum;

  const int b   = blockIdx.x;
  const int tid = threadIdx.x;
  for (int i = tid; i < PP; i += 256)
    vals[i] = conf_masked[(size_t)b * PP + i];
  if (tid == 0) {
    s_prefix = 0u;
    s_krem   = (unsigned)(NPR * pos_count[b]);
    s_sum    = 0.f;
  }
  __syncthreads();
  if (s_krem == 0) return;  // uniform exit

  for (int round = 0; round < 4; ++round) {
    const int shift = 24 - 8 * round;
    hist[tid] = 0u; hsum[tid] = 0.f;
    __syncthreads();
    const unsigned pref  = s_prefix;
    const unsigned pmask = (round == 0) ? 0u : (0xFFFFFFFFu << (shift + 8));
    for (int i = tid; i < PP; i += 256) {
      const float x = vals[i];
      const unsigned bts = __float_as_uint(x);
      const unsigned key = (bts & 0x80000000u) ? ~bts : (bts | 0x80000000u);
      if ((key & pmask) == (pref & pmask)) {
        const unsigned byte = (key >> shift) & 255u;
        atomicAdd(&hist[byte], 1u);
        atomicAdd(&hsum[byte], x);
      }
    }
    __syncthreads();
    if (tid == 0) {
      unsigned krem = s_krem;
      float ss = s_sum;
      int bin = 255;
      while (bin > 0 && krem > hist[bin]) {   // bins fully above threshold: take all
        krem -= hist[bin];
        ss   += hsum[bin];
        --bin;
      }
      s_prefix = pref | ((unsigned)bin << shift);
      s_krem = krem;
      s_sum = ss;
    }
    __syncthreads();
  }
  if (tid == 0) {
    const unsigned key = s_prefix;
    const unsigned bts = (key & 0x80000000u) ? (key & 0x7FFFFFFFu) : ~key;
    const float thr = __uint_as_float(bts);
    atomicAdd(&accum[2], s_sum + (float)s_krem * thr);  // krem ties at threshold value
  }
}

__global__ void finalize_kernel(const int* __restrict__ pos_count,
                                const float* __restrict__ accum,
                                float* __restrict__ out)
{
  if (threadIdx.x == 0) {
    int tp = 0;
    for (int i = 0; i < BB; ++i) tp += pos_count[i];
    const float ftp = (float)tp;
    out[0] = (tp != 0) ? accum[0] / ftp : 0.f;                 // loc_loss (alpha=1)
    out[1] = (tp != 0) ? (accum[1] + accum[2]) / ftp : 0.f;    // conf_loss
  }
}

extern "C" void kernel_launch(void* const* d_in, const int* in_sizes, int n_in,
                              void* d_out, int out_size, void* d_ws, size_t ws_size,
                              hipStream_t stream) {
  const float* act_deltas  = (const float*)d_in[0];  // [B,P,4]
  const float* act_labels  = (const float*)d_in[1];  // [B,P,C]
  const float* pred_deltas = (const float*)d_in[2];  // [B,P,4]
  const float* pred_labels = (const float*)d_in[3];  // [B,P,C]
  float* out = (float*)d_out;                        // 2 floats

  float* conf_masked = (float*)d_ws;
  int*   pos_count   = (int*)(conf_masked + (size_t)BB * PP);
  float* accum       = (float*)(pos_count + BB);

  init_kernel<<<1, 64, 0, stream>>>(pos_count, accum);
  dim3 g1(TILES, BB);
  conf_kernel<<<g1, 32, 0, stream>>>(act_deltas, act_labels, pred_deltas,
                                     pred_labels, conf_masked, pos_count, accum);
  mine_kernel<<<BB, 256, 0, stream>>>(conf_masked, pos_count, accum);
  finalize_kernel<<<1, 32, 0, stream>>>(pos_count, accum, out);
}
